// TemporalChannelAggregator_37512244364116
// MI455X (gfx1250) — matrix operand order repaired and checked by reading, not compiled, not run on hardware
//
#include <hip/hip_runtime.h>
#include <hip/hip_bf16.h>
#include <stdint.h>
#include <stddef.h>

typedef __bf16 bf16_t;
typedef __attribute__((ext_vector_type(16))) __bf16 v16bf;
typedef __attribute__((ext_vector_type(8)))  float  v8f;

#define T_MAX  2048
#define EMBED  512
#define NVARS  64
#define GRUH   64
#define G3     192   // 3 * GRUH

// ---- workspace layout (bytes) ----
// wih_bf : 192*512*2 = 196608
// whh_bf : 192*64*2  = 24576   -> 221184 (256-aligned)
// xg     : 2048*64*192*4 = 100663296 -> 100884480
// hT     : 64*64*4
static constexpr size_t WIH_BF_OFF = 0;
static constexpr size_t WHH_BF_OFF = 196608;
static constexpr size_t XG_OFF     = 221184;
static constexpr size_t HT_OFF     = 100884480;

__device__ __forceinline__ float sigmoidf_(float x) {
    return 1.0f / (1.0f + __expf(-x));
}

// ---------------------------------------------------------------------------
// B fragment (bf16, 32x16 slice of k-major-contiguous [N][K] weight array):
// lane L: n = L&15, khalf = L>>4; element e of the v16bf <-> k = k0 + khalf*16 + e
// => 16 contiguous bf16 per lane = two 16B loads.
// ---------------------------------------------------------------------------
__device__ __forceinline__ v16bf load_b_frag(const bf16_t* __restrict__ base,
                                             int row_stride, int nglob,
                                             int k0, int khalf) {
    const bf16_t* p = base + (size_t)nglob * row_stride + k0 + khalf * 16;
    union { v16bf v; uint4 q[2]; } u;
    u.q[0] = *(const uint4*)(p);
    u.q[1] = *(const uint4*)(p + 8);
    return u.v;
}

// ---------------------------------------------------------------------------
// A fragment (16-bit 16x32, ISA 7.12.2): lane L: m = L&15, khalf = L>>4.
// dword j : k = 16*khalf + 2j (contiguous, matches load_b_frag).
// Built from fp32 pairs (8-byte loads) + cvt to bf16. Works for global or LDS rows.
// ---------------------------------------------------------------------------
__device__ __forceinline__ v16bf make_a_frag_f32(const float* __restrict__ row,
                                                 int k0, int khalf) {
    v16bf a;
#pragma unroll
    for (int j = 0; j < 8; ++j) {
        int kb = 16 * khalf + 2 * j;
        float2 f = *(const float2*)(row + k0 + kb);
        a[2 * j]     = (bf16_t)f.x;
        a[2 * j + 1] = (bf16_t)f.y;
    }
    return a;
}

// ---------------------------------------------------------------------------
// Kernel 0: convert GRU weights fp32 -> bf16 (same [N][K] row-major layout).
// ---------------------------------------------------------------------------
__global__ void cvt_weights_kernel(const float* __restrict__ W_ih,
                                   const float* __restrict__ W_hh,
                                   bf16_t* __restrict__ wih_bf,
                                   bf16_t* __restrict__ whh_bf) {
    int i = blockIdx.x * blockDim.x + threadIdx.x;
    if (i < G3 * EMBED) wih_bf[i] = (bf16_t)W_ih[i];
    if (i < G3 * GRUH)  whh_bf[i] = (bf16_t)W_hh[i];
}

// ---------------------------------------------------------------------------
// Kernel 1: x_gates = emb @ W_ih^T + b_ih   ([131072,512] x [512,192])
// One wave per 16-row M-tile, all 12 N-tiles in 12 fp32 accumulators,
// K-loop of 16 (K=512, 32 per WMMA). HBM-bound by design.
// Output layout: xg[t][v][g] (matches reference 'tvg').
// ---------------------------------------------------------------------------
__global__ void __launch_bounds__(128)
xgates_gemm_kernel(const float* __restrict__ emb,
                   const bf16_t* __restrict__ wih_bf,
                   const float* __restrict__ b_ih,
                   float* __restrict__ xg) {
    const int lane  = threadIdx.x & 31;
    const int wave  = threadIdx.x >> 5;
    const int khalf = lane >> 4;
    const int n     = lane & 15;
    const int mtile = blockIdx.x * 4 + wave;

    // A row for this lane: global row R = mtile*16 + (lane&15); emb is [V][T][E] flat.
    const float* arow = emb + ((size_t)mtile * 16 + (lane & 15)) * EMBED;

    float bias[12];
#pragma unroll
    for (int nt = 0; nt < 12; ++nt) bias[nt] = b_ih[nt * 16 + n];

    v8f acc[12] = {};

    for (int kt = 0; kt < EMBED / 32; ++kt) {
        const int k0 = kt * 32;
        v16bf a = make_a_frag_f32(arow, k0, khalf);
#pragma unroll
        for (int nt = 0; nt < 12; ++nt) {
            v16bf b = load_b_frag(wih_bf, EMBED, nt * 16 + n, k0, khalf);
            acc[nt] = __builtin_amdgcn_wmma_f32_16x16x32_bf16(
                false, a, false, b, (short)0, acc[nt], false, false);
        }
    }

    // C layout: lane holds (m = r + 8*khalf, n); row R -> (v = R>>11, t = R&2047)
#pragma unroll
    for (int nt = 0; nt < 12; ++nt) {
#pragma unroll
        for (int r = 0; r < 8; ++r) {
            int R = mtile * 16 + r + 8 * khalf;
            int v = R >> 11;
            int t = R & (T_MAX - 1);
            xg[((size_t)t * NVARS + v) * G3 + nt * 16 + n] = acc[nt][r] + bias[nt];
        }
    }
}

// ---------------------------------------------------------------------------
// Kernel 2: sequential GRU scan. 4 workgroups x 16 variables, 4 waves each.
// Wave w owns hidden columns [16w,16w+16): gates r/z/n for that slab.
// h kept fp32 in LDS (A-source) + old-h slab in registers (C layout).
// W_hh B-fragments preloaded once (6 x 8 VGPRs).
// ---------------------------------------------------------------------------
__global__ void __launch_bounds__(128)
gru_scan_kernel(const float* __restrict__ xg,
                const bf16_t* __restrict__ whh_bf,
                const float* __restrict__ b_hh,
                const int* __restrict__ lengths,
                float* __restrict__ hT) {
    __shared__ float h32[16][66];   // [var][hidden], padded stride vs 64 banks

    const int lane  = threadIdx.x & 31;
    const int w     = threadIdx.x >> 5;
    const int khalf = lane >> 4;
    const int n     = lane & 15;
    const int m     = lane & 15;            // A-fragment row (variable index)
    const int vbase = blockIdx.x * 16;

    // Preload W_hh fragments: gate g in {r,z,n}, ktile in {0,1} (K = GRUH = 64)
    v16bf bfr[3][2];
#pragma unroll
    for (int g = 0; g < 3; ++g)
#pragma unroll
        for (int kt = 0; kt < 2; ++kt)
            bfr[g][kt] = load_b_frag(whh_bf, GRUH, g * 64 + 16 * w + n, kt * 32, khalf);

    const float bhr = b_hh[0 * 64 + 16 * w + n];
    const float bhz = b_hh[1 * 64 + 16 * w + n];
    const float bhn = b_hh[2 * 64 + 16 * w + n];

    int len_r[8];
#pragma unroll
    for (int r = 0; r < 8; ++r) len_r[r] = lengths[vbase + r + 8 * khalf];

    float hreg[8];
#pragma unroll
    for (int r = 0; r < 8; ++r) hreg[r] = 0.0f;

    // zero the LDS h (each wave zeroes its own column slab)
#pragma unroll
    for (int r = 0; r < 8; ++r) h32[r + 8 * khalf][16 * w + n] = 0.0f;
    __syncthreads();

    for (int t = 0; t < T_MAX; ++t) {
        // x-gate inputs for this wave's slab (g = 16w+n within each gate block)
        float xr[8], xz[8], xn[8];
#pragma unroll
        for (int r = 0; r < 8; ++r) {
            size_t base = ((size_t)t * NVARS + vbase + r + 8 * khalf) * G3 + 16 * w + n;
            xr[r] = xg[base];
            xz[r] = xg[base + 64];
            xn[r] = xg[base + 128];
        }

        // A fragments (h, bf16) from LDS
        v16bf a0 = make_a_frag_f32(&h32[m][0], 0, khalf);
        v16bf a1 = make_a_frag_f32(&h32[m][0], 32, khalf);
        __syncthreads();   // all waves done reading h32 before it is overwritten

        v8f ar = {}, az = {}, an = {};
        ar = __builtin_amdgcn_wmma_f32_16x16x32_bf16(false, a0, false, bfr[0][0], (short)0, ar, false, false);
        ar = __builtin_amdgcn_wmma_f32_16x16x32_bf16(false, a1, false, bfr[0][1], (short)0, ar, false, false);
        az = __builtin_amdgcn_wmma_f32_16x16x32_bf16(false, a0, false, bfr[1][0], (short)0, az, false, false);
        az = __builtin_amdgcn_wmma_f32_16x16x32_bf16(false, a1, false, bfr[1][1], (short)0, az, false, false);
        an = __builtin_amdgcn_wmma_f32_16x16x32_bf16(false, a0, false, bfr[2][0], (short)0, an, false, false);
        an = __builtin_amdgcn_wmma_f32_16x16x32_bf16(false, a1, false, bfr[2][1], (short)0, an, false, false);

#pragma unroll
        for (int r = 0; r < 8; ++r) {
            float rg = sigmoidf_(xr[r] + ar[r] + bhr);
            float zg = sigmoidf_(xz[r] + az[r] + bhz);
            float ng = tanhf(xn[r] + rg * (an[r] + bhn));
            float hn = (1.0f - zg) * ng + zg * hreg[r];
            hreg[r]  = (t < len_r[r]) ? hn : hreg[r];   // per-variable freeze
            h32[r + 8 * khalf][16 * w + n] = hreg[r];
        }
        __syncthreads();
    }

    // final hidden state: hT[v][h]
#pragma unroll
    for (int r = 0; r < 8; ++r)
        hT[(size_t)(vbase + r + 8 * khalf) * GRUH + 16 * w + n] = hreg[r];
}

// ---------------------------------------------------------------------------
// Kernel 3: tiny MLP head: out = W2 @ relu(W1 @ hT.flat + b1) + b2
// ---------------------------------------------------------------------------
__global__ void __launch_bounds__(64)
mlp_kernel(const float* __restrict__ hT,
           const float* __restrict__ W1, const float* __restrict__ b1,
           const float* __restrict__ W2, const float* __restrict__ b2,
           float* __restrict__ out) {
    __shared__ float hid[64];
    const int j = threadIdx.x;
    float acc = b1[j];
    const float* wrow = W1 + (size_t)j * (NVARS * GRUH);
    for (int i = 0; i < NVARS * GRUH; ++i) acc += wrow[i] * hT[i];
    hid[j] = fmaxf(acc, 0.0f);
    __syncthreads();
    if (j == 0) {
        float o = b2[0];
        for (int k = 0; k < 64; ++k) o += W2[k] * hid[k];
        out[0] = o;
    }
}

// ---------------------------------------------------------------------------
extern "C" void kernel_launch(void* const* d_in, const int* in_sizes, int n_in,
                              void* d_out, int out_size, void* d_ws, size_t ws_size,
                              hipStream_t stream) {
    (void)in_sizes; (void)n_in; (void)out_size; (void)ws_size;

    const float* emb  = (const float*)d_in[0];
    const int*   lens = (const int*)d_in[1];
    const float* W_ih = (const float*)d_in[2];
    const float* W_hh = (const float*)d_in[3];
    const float* b_ih = (const float*)d_in[4];
    const float* b_hh = (const float*)d_in[5];
    const float* W1   = (const float*)d_in[6];
    const float* b1   = (const float*)d_in[7];
    const float* W2   = (const float*)d_in[8];
    const float* b2   = (const float*)d_in[9];
    float* out = (float*)d_out;

    char* ws = (char*)d_ws;
    bf16_t* wih_bf = (bf16_t*)(ws + WIH_BF_OFF);
    bf16_t* whh_bf = (bf16_t*)(ws + WHH_BF_OFF);
    float*  xg     = (float*)(ws + XG_OFF);
    float*  hT     = (float*)(ws + HT_OFF);

    cvt_weights_kernel<<<(G3 * EMBED + 255) / 256, 256, 0, stream>>>(W_ih, W_hh, wih_bf, whh_bf);
    xgates_gemm_kernel<<<(NVARS * T_MAX) / 16 / 4, 128, 0, stream>>>(emb, wih_bf, b_ih, xg);
    gru_scan_kernel<<<NVARS / 16, 128, 0, stream>>>(xg, whh_bf, b_hh, lens, hT);
    mlp_kernel<<<1, 64, 0, stream>>>(hT, W1, b1, W2, b2, out);
}
